// ScattterAttentionLayer_mul_a_69337952026836
// MI455X (gfx1250) — matrix operand order repaired and checked by reading, not compiled, not run on hardware
//
#include <hip/hip_runtime.h>

typedef __attribute__((ext_vector_type(16))) _Float16 v16h;
typedef __attribute__((ext_vector_type(8)))  float    v8f;

#define IN_F  256
#define OUT_F 64

// ---------------------------------------------------------------------------
// Pack W (fp32, [256,64] row-major) into f16 WMMA B-fragment order.
// B frag (16x16x32 f16, wave32): lane l -> n = l%16, g = l/16; vector elem e
// holds B[k][n] with k = g*16 + e (contiguous k order, same as the A frag).
// Bpack flat index: ((kc*4 + nt)*32 + lane)*16 + e   (kc = K-chunk, nt = N-tile)
// ---------------------------------------------------------------------------
__global__ void pack_W_kernel(const float* __restrict__ W, _Float16* __restrict__ Bpack) {
    int t = blockIdx.x * blockDim.x + threadIdx.x;
    if (t >= 8 * 4 * 32 * 16) return;
    int e    = t & 15;
    int lane = (t >> 4) & 31;
    int nt   = (t >> 9) & 3;
    int kc   = t >> 11;
    int g = lane >> 4;
    int n = lane & 15;
    int k = kc * 32 + g * 16 + e;
    Bpack[t] = (_Float16)W[k * OUT_F + nt * 16 + n];
}

// ---------------------------------------------------------------------------
// s0 = x @ W via v_wmma_f32_16x16x32_f16.  One wave computes a 16x64 tile:
// 4 N-tiles x 8 K-chunks of 32.  A frag, same k order as B: lane group g holds
// K[g*16..g*16+16) in elems 0..15, row = lane%16.
// ---------------------------------------------------------------------------
__global__ void gemm_s0_kernel(const float* __restrict__ x,
                               const _Float16* __restrict__ Bpack,
                               float* __restrict__ s0, int N) {
    int wave = (int)((blockIdx.x * blockDim.x + threadIdx.x) >> 5);
    int lane = threadIdx.x & 31;
    int Mtiles = (N + 15) >> 4;
    if (wave >= Mtiles) return;

    int g = lane >> 4;        // lane group (K half select)
    int n = lane & 15;        // A row within tile / B & D column within tile
    int row = wave * 16 + n;
    int rowc = row < N ? row : N - 1;
    const float*  xrow = x + (size_t)rowc * IN_F;
    const v16h*   bp   = (const v16h*)Bpack;

    v8f acc0 = {}, acc1 = {}, acc2 = {}, acc3 = {};

#pragma unroll
    for (int kc = 0; kc < 8; ++kc) {
        int kb = kc * 32;
        const float4* p0 = (const float4*)(xrow + kb + g * 16);
        const float4* p1 = (const float4*)(xrow + kb + g * 16 + 8);
        float4 f0 = p0[0], f1 = p0[1];
        float4 f2 = p1[0], f3 = p1[1];
        v16h av;
        av[0]  = (_Float16)f0.x; av[1]  = (_Float16)f0.y;
        av[2]  = (_Float16)f0.z; av[3]  = (_Float16)f0.w;
        av[4]  = (_Float16)f1.x; av[5]  = (_Float16)f1.y;
        av[6]  = (_Float16)f1.z; av[7]  = (_Float16)f1.w;
        av[8]  = (_Float16)f2.x; av[9]  = (_Float16)f2.y;
        av[10] = (_Float16)f2.z; av[11] = (_Float16)f2.w;
        av[12] = (_Float16)f3.x; av[13] = (_Float16)f3.y;
        av[14] = (_Float16)f3.z; av[15] = (_Float16)f3.w;

        v16h b0 = bp[(kc * 4 + 0) * 32 + lane];
        v16h b1 = bp[(kc * 4 + 1) * 32 + lane];
        v16h b2 = bp[(kc * 4 + 2) * 32 + lane];
        v16h b3 = bp[(kc * 4 + 3) * 32 + lane];

        acc0 = __builtin_amdgcn_wmma_f32_16x16x32_f16(false, av, false, b0, (short)0, acc0, false, false);
        acc1 = __builtin_amdgcn_wmma_f32_16x16x32_f16(false, av, false, b1, (short)0, acc1, false, false);
        acc2 = __builtin_amdgcn_wmma_f32_16x16x32_f16(false, av, false, b2, (short)0, acc2, false, false);
        acc3 = __builtin_amdgcn_wmma_f32_16x16x32_f16(false, av, false, b3, (short)0, acc3, false, false);
    }

    // D layout: VGPR j -> row = tile*16 + g*8 + j, col = nt*16 + (lane%16)
#pragma unroll
    for (int j = 0; j < 8; ++j) {
        int r = wave * 16 + g * 8 + j;
        if (r < N) {
            float* out = s0 + (size_t)r * OUT_F + n;
            out[0]  = acc0[j];
            out[16] = acc1[j];
            out[32] = acc2[j];
            out[48] = acc3[j];
        }
    }
}

__global__ void zero_kernel(float* __restrict__ p, long n) {
    long i = (long)blockIdx.x * blockDim.x + threadIdx.x;
    long stride = (long)gridDim.x * blockDim.x;
    for (; i < n; i += stride) p[i] = 0.0f;
}

// ---------------------------------------------------------------------------
// COO SpMM: 16 threads per edge, each handling a float4 feature chunk so the
// gather of src[c] and atomic scatter to dst[r] are each one coalesced 256B row.
// ---------------------------------------------------------------------------
__global__ void spmm_atomic_kernel(const int* __restrict__ rows,
                                   const int* __restrict__ cols,
                                   const float* __restrict__ val,
                                   const float* __restrict__ src,
                                   float* __restrict__ dst, int E) {
    long t = (long)blockIdx.x * blockDim.x + threadIdx.x;
    if (t >= (long)E * 16) return;
    int e = (int)(t >> 4);
    int f = ((int)t & 15) * 4;
    int r = rows[e];
    int c = cols[e];
    float v = val[e];
    const float4 sv = *(const float4*)(src + (size_t)c * OUT_F + f);
    float* d = dst + (size_t)r * OUT_F + f;
    atomicAdd(d + 0, v * sv.x);
    atomicAdd(d + 1, v * sv.y);
    atomicAdd(d + 2, v * sv.z);
    atomicAdd(d + 3, v * sv.w);
}

// ---------------------------------------------------------------------------
// Fused attention epilogue.  One wave per node r.
// e[r][i] = dot(concat(rowA,rowB), a[i]) where rows come from s0 (r < N/2)
// or channel i (abs for i>=3).  Then softmax, then the permuted channel mix.
// H = 6 contiguous [N,64] channel buffers (hA,hA2,hA3,hs1,hs2,hs3).
// ---------------------------------------------------------------------------
__global__ void attention_kernel(const float* __restrict__ s0,
                                 const float* __restrict__ H,
                                 const float* __restrict__ a,
                                 float* __restrict__ hprime,
                                 float* __restrict__ att_out, int N) {
    int wib  = threadIdx.x >> 5;
    int lane = threadIdx.x & 31;
    int r = blockIdx.x * 8 + wib;
    if (r >= N) return;
    size_t chanStride = (size_t)N * OUT_F;

    float evals[6];
    if (2 * r < N) {
        int rowA = 2 * r, rowB = 2 * r + 1;
        float x0 = s0[(size_t)rowA * 64 + lane];
        float x1 = s0[(size_t)rowA * 64 + 32 + lane];
        float x2 = s0[(size_t)rowB * 64 + lane];
        float x3 = s0[(size_t)rowB * 64 + 32 + lane];
#pragma unroll
        for (int i = 0; i < 6; ++i) {
            const float* ai = a + i * 128;
            float p = x0 * ai[lane] + x1 * ai[32 + lane]
                    + x2 * ai[64 + lane] + x3 * ai[96 + lane];
#pragma unroll
            for (int off = 16; off > 0; off >>= 1) p += __shfl_xor(p, off, 32);
            evals[i] = p;
        }
    } else {
        int rowA = 2 * r - N, rowB = rowA + 1;
#pragma unroll
        for (int i = 0; i < 6; ++i) {
            const float* C = H + (size_t)i * chanStride;
            float x0 = C[(size_t)rowA * 64 + lane];
            float x1 = C[(size_t)rowA * 64 + 32 + lane];
            float x2 = C[(size_t)rowB * 64 + lane];
            float x3 = C[(size_t)rowB * 64 + 32 + lane];
            if (i >= 3) { x0 = fabsf(x0); x1 = fabsf(x1); x2 = fabsf(x2); x3 = fabsf(x3); }
            const float* ai = a + i * 128;
            float p = x0 * ai[lane] + x1 * ai[32 + lane]
                    + x2 * ai[64 + lane] + x3 * ai[96 + lane];
#pragma unroll
            for (int off = 16; off > 0; off >>= 1) p += __shfl_xor(p, off, 32);
            evals[i] = p;
        }
    }

    float m = evals[0];
#pragma unroll
    for (int i = 1; i < 6; ++i) m = fmaxf(m, evals[i]);
    float att[6], den = 0.0f;
#pragma unroll
    for (int i = 0; i < 6; ++i) { att[i] = __expf(evals[i] - m); den += att[i]; }
    float inv = 1.0f / den;
#pragma unroll
    for (int i = 0; i < 6; ++i) att[i] *= inv;

    // h_prime[r][k] = (1/6) * sum_j att[j] * chan[(64j+k)%6][r][(64j+k)/6]
#pragma unroll
    for (int half = 0; half < 2; ++half) {
        int k = lane + half * 32;
        float s = 0.0f;
#pragma unroll
        for (int j = 0; j < 6; ++j) {
            int idx = j * 64 + k;
            int c = idx % 6;
            int f = idx / 6;
            float v = H[(size_t)c * chanStride + (size_t)r * 64 + f];
            if (c >= 3) v = fabsf(v);
            s += att[j] * v;
        }
        hprime[(size_t)r * 64 + k] = s * (1.0f / 6.0f);
    }

    if (lane < 6) {
        float v = att[0];
        if (lane == 1) v = att[1];
        if (lane == 2) v = att[2];
        if (lane == 3) v = att[3];
        if (lane == 4) v = att[4];
        if (lane == 5) v = att[5];
        att_out[(size_t)r * 6 + lane] = v;
    }
}

extern "C" void kernel_launch(void* const* d_in, const int* in_sizes, int n_in,
                              void* d_out, int out_size, void* d_ws, size_t ws_size,
                              hipStream_t stream) {
    const float* x      = (const float*)d_in[0];
    const float* W      = (const float*)d_in[1];
    const float* a      = (const float*)d_in[2];
    const int*   A_idx  = (const int*)d_in[3];
    const float* A_val  = (const float*)d_in[4];
    const int*   P1_idx = (const int*)d_in[5];
    const float* P1_val = (const float*)d_in[6];
    const int*   P2_idx = (const int*)d_in[7];
    const float* P2_val = (const float*)d_in[8];
    const int*   P3_idx = (const int*)d_in[9];
    const float* P3_val = (const float*)d_in[10];

    const int N = in_sizes[0] / IN_F;
    const int E = in_sizes[3] / 2;

    char* ws = (char*)d_ws;
    _Float16* Bpack = (_Float16*)ws;                         // 32 KB packed W
    float* s0 = (float*)(ws + 64 * 1024);                    // [N,64]
    float* H  = s0 + (size_t)N * OUT_F;                      // 6 x [N,64] channels
    float* hA  = H + 0 * (size_t)N * OUT_F;
    float* hA2 = H + 1 * (size_t)N * OUT_F;
    float* hA3 = H + 2 * (size_t)N * OUT_F;
    float* hs1 = H + 3 * (size_t)N * OUT_F;
    float* hs2 = H + 4 * (size_t)N * OUT_F;
    float* hs3 = H + 5 * (size_t)N * OUT_F;

    float* hprime  = (float*)d_out;
    float* att_out = hprime + (size_t)N * OUT_F;

    pack_W_kernel<<<64, 256, 0, stream>>>(W, Bpack);

    int Mtiles = (N + 15) / 16;
    int gblocks = (Mtiles + 7) / 8;                          // 8 waves / block
    gemm_s0_kernel<<<gblocks, 256, 0, stream>>>(x, Bpack, s0, N);

    zero_kernel<<<2048, 256, 0, stream>>>(H, 6L * N * OUT_F);

    long sp_threads = (long)E * 16;
    int sp_blocks = (int)((sp_threads + 255) / 256);
    spmm_atomic_kernel<<<sp_blocks, 256, 0, stream>>>(A_idx,  A_idx  + E, A_val,  s0,  hA,  E);
    spmm_atomic_kernel<<<sp_blocks, 256, 0, stream>>>(A_idx,  A_idx  + E, A_val,  hA,  hA2, E);
    spmm_atomic_kernel<<<sp_blocks, 256, 0, stream>>>(A_idx,  A_idx  + E, A_val,  hA2, hA3, E);
    spmm_atomic_kernel<<<sp_blocks, 256, 0, stream>>>(P1_idx, P1_idx + E, P1_val, s0,  hs1, E);
    spmm_atomic_kernel<<<sp_blocks, 256, 0, stream>>>(P2_idx, P2_idx + E, P2_val, s0,  hs2, E);
    spmm_atomic_kernel<<<sp_blocks, 256, 0, stream>>>(P3_idx, P3_idx + E, P3_val, s0,  hs3, E);

    attention_kernel<<<(N + 7) / 8, 256, 0, stream>>>(s0, H, a, hprime, att_out, N);
}
